// MultiHeadSelfAttention_5282809774118
// MI455X (gfx1250) — matrix operand order repaired and checked by reading, not compiled, not run on hardware
//
#include <hip/hip_runtime.h>

typedef __bf16 bf16;
typedef __attribute__((ext_vector_type(16))) __bf16 v16bf;
typedef __attribute__((ext_vector_type(8)))  __bf16 v8bf;
typedef __attribute__((ext_vector_type(8)))  float  v8f;

namespace {
constexpr int kB  = 4;
constexpr int kN  = 4096;   // queries per batch (64*64)
constexpr int kC  = 512;
constexpr int kNH = 8;
constexpr int kHD = 64;
constexpr int kNk = 1024;   // keys per batch (32*32)
constexpr int kKc = 2048;   // im2col K for the 2x2 stride-2 conv
constexpr float kScale = 0.125f; // 1/sqrt(64)
enum { EPI_F32 = 0, EPI_HEADS = 2, EPI_HEADS_T = 3 };
constexpr int kBtPad = 40;  // LDS row stride (elements) for transposed B tile
}

__device__ __forceinline__ v8f wmma_bf16(v16bf a, v16bf b, v8f c) {
  return __builtin_amdgcn_wmma_f32_16x16x32_bf16(false, a, false, b, (short)0, c,
                                                 false, false);
}

// A fragment, 16x32 bf16, A row-major [M x K].
// Lanes 0-15: rows 0-15, K in {0..15}; lanes 16-31: K in {16..31}.
__device__ __forceinline__ v16bf load_a_rm(const bf16* __restrict__ A, int lda,
                                           int m0, int k0, int lane) {
  int row = m0 + (lane & 15);
  int kb  = k0 + ((lane & 16) ? 16 : 0);
  const bf16* p = A + (size_t)row * lda + kb;
  v8bf lo = *(const v8bf*)p;
  v8bf hi = *(const v8bf*)(p + 8);
  v16bf a;
#pragma unroll
  for (int j = 0; j < 8; ++j) { a[j] = lo[j]; a[8 + j] = hi[j]; }
  return a;
}

// B fragment where logical B[k][n] = Bt[n][k] (Bt row-major [N x K]) -> two
// contiguous 16B loads per lane (works for global or LDS pointers).
__device__ __forceinline__ v16bf load_b_tn(const bf16* Bt, int ldbt,
                                           int k0, int n0, int lane) {
  int col = n0 + (lane & 15);
  int kb  = k0 + ((lane & 16) ? 16 : 0);
  const bf16* p = Bt + (size_t)col * ldbt + kb;
  v8bf lo = *(const v8bf*)p;
  v8bf hi = *(const v8bf*)(p + 8);
  v16bf b;
#pragma unroll
  for (int j = 0; j < 8; ++j) { b[j] = lo[j]; b[8 + j] = hi[j]; }
  return b;
}

__global__ __launch_bounds__(256) void cvt_f2bf(const float* __restrict__ in,
                                                bf16* __restrict__ out, int n) {
  int i = blockIdx.x * 256 + threadIdx.x;
  if (i < n) out[i] = (bf16)in[i];
}

// im2col for kernel=stride=2 conv: A[(b,oy,ox)][(ky,kx,ci)] bf16
__global__ __launch_bounds__(256) void im2col_bf16(const float* __restrict__ x,
                                                   bf16* __restrict__ out) {
  size_t i = (size_t)blockIdx.x * 256 + threadIdx.x;
  if (i >= (size_t)kB * kNk * kKc) return;
  int cc = (int)(i & (kKc - 1));
  size_t r = i >> 11;
  int b = (int)(r >> 10);
  int p = (int)(r & 1023);
  int oy = p >> 5, ox = p & 31;
  int kk = cc >> 9, ci = cc & 511;
  int ky = kk >> 1, kx = kk & 1;
  int iy = oy * 2 + ky, ix = ox * 2 + kx;
  out[i] = (bf16)x[((size_t)b * kN + iy * 64 + ix) * kC + ci];
}

// rows = kB*kNk, width kC. One block per row.
__global__ __launch_bounds__(256) void layernorm_bf16(const float* __restrict__ in,
                                                      const float* __restrict__ g,
                                                      const float* __restrict__ bta,
                                                      bf16* __restrict__ out) {
  int row = blockIdx.x, t = threadIdx.x;
  const float* p = in + (size_t)row * kC;
  float v0 = p[t], v1 = p[t + 256];
  __shared__ float red[256];
  red[t] = v0 + v1;
  __syncthreads();
  for (int s = 128; s > 0; s >>= 1) { if (t < s) red[t] += red[t + s]; __syncthreads(); }
  float mu = red[0] * (1.0f / kC);
  __syncthreads();
  float d0 = v0 - mu, d1 = v1 - mu;
  red[t] = d0 * d0 + d1 * d1;
  __syncthreads();
  for (int s = 128; s > 0; s >>= 1) { if (t < s) red[t] += red[t + s]; __syncthreads(); }
  float rstd = rsqrtf(red[0] * (1.0f / kC) + 1e-6f);
  bf16* o = out + (size_t)row * kC;
  o[t]       = (bf16)(d0 * rstd * g[t]       + bta[t]);
  o[t + 256] = (bf16)(d1 * rstd * g[t + 256] + bta[t + 256]);
}

// Generic bf16 WMMA GEMM: C = A[MxK] * B[KxN] + bias.
// 256 threads = 8 waves as 4(m)x2(n); each wave owns a 32x32 tile (2x2 WMMAs).
// Block tile 128x64. B k-chunk staged to LDS transposed with coalesced b128
// loads so fragment reads are contiguous ds_load_b128.
__global__ __launch_bounds__(256) void gemm_bf16_wmma(
    const bf16* __restrict__ A, const bf16* __restrict__ Bm,
    const float* __restrict__ bias, float* __restrict__ outF,
    bf16* __restrict__ outB, int M, int N, int K, int mode, int lg_rows) {
  __shared__ bf16 Bt[64][kBtPad];  // [n_local][k], padded row stride

  int t = threadIdx.x;
  int w = t >> 5, lane = t & 31;
  int m0 = blockIdx.y * 128 + (w & 3) * 32;
  int nblk = blockIdx.x * 64;
  int n0 = nblk + (w >> 2) * 32;
  int nloc = (w >> 2) * 32;

  // cooperative B stage indices: 32 rows(k) x 64 cols(n), 8 threads per row
  int sk = t >> 3;            // 0..31
  int sn = (t & 7) * 8;       // 0,8,..,56

  v8f c00 = {}, c01 = {}, c10 = {}, c11 = {};
  for (int k0 = 0; k0 < K; k0 += 32) {
    __syncthreads();
    {
      v8bf vb = *(const v8bf*)(Bm + (size_t)(k0 + sk) * N + nblk + sn);
#pragma unroll
      for (int i = 0; i < 8; ++i) Bt[sn + i][sk] = vb[i];
    }
    // speculative prefetch of next A chunk (silently dropped if OOB)
    __builtin_prefetch(A + (size_t)(m0 + (lane & 15)) * K + k0 + 32, 0, 1);
    __syncthreads();

    v16bf a0 = load_a_rm(A, K, m0, k0, lane);
    v16bf a1 = load_a_rm(A, K, m0 + 16, k0, lane);
    v16bf b0 = load_b_tn(&Bt[0][0], kBtPad, 0, nloc, lane);
    v16bf b1 = load_b_tn(&Bt[0][0], kBtPad, 0, nloc + 16, lane);
    c00 = wmma_bf16(a0, b0, c00);
    c01 = wmma_bf16(a0, b1, c01);
    c10 = wmma_bf16(a1, b0, c10);
    c11 = wmma_bf16(a1, b1, c11);
  }

  // C layout: elem r -> row tile_m + r + (lane>=16 ? 8 : 0), col tile_n + (lane&15)
  int ri = (lane & 16) ? 8 : 0;
  int cb = lane & 15;
#pragma unroll
  for (int ti = 0; ti < 2; ++ti) {
#pragma unroll
    for (int tj = 0; tj < 2; ++tj) {
      const v8f& c = (ti == 0) ? (tj == 0 ? c00 : c01) : (tj == 0 ? c10 : c11);
      int col = n0 + tj * 16 + cb;
      float bv = bias[col];
#pragma unroll
      for (int r = 0; r < 8; ++r) {
        int row = m0 + ti * 16 + ri + r;
        float v = c[r] + bv;
        if (mode == EPI_F32) {
          outF[(size_t)row * N + col] = v;
        } else {
          // channel col -> (h = col%8, d = col/8); row -> (batch bb, token n)
          int bb = row >> lg_rows;
          int n = row & ((1 << lg_rows) - 1);
          int h = col & 7, d = col >> 3;
          if (mode == EPI_HEADS) {
            // [bh][n][d]
            outB[(((size_t)(bb * kNH + h)) << lg_rows | n) * kHD + d] = (bf16)v;
          } else {
            // EPI_HEADS_T: [bh][d][n]  (for V: contiguous along keys)
            outB[(((size_t)(bb * kNH + h) * kHD + d) << lg_rows) | n] = (bf16)v;
          }
        }
      }
    }
  }
}

// One block per (b, h, 16-query tile). Online softmax over 2 chunks of 512 keys.
// K is [bh][nk][d]; V is stored TRANSPOSED [bh][d][nk] so both WMMA B-operands
// are contiguous two-b128 loads per lane.
__global__ __launch_bounds__(256) void attn_wmma(
    const bf16* __restrict__ qh, const bf16* __restrict__ kh,
    const bf16* __restrict__ vt, bf16* __restrict__ out_attn) {
  int bh = blockIdx.y;
  int b = bh >> 3, h = bh & 7;
  int q0 = blockIdx.x * 16;
  const bf16* Q  = qh + ((size_t)bh * kN + q0) * kHD;  // [16 x 64]
  const bf16* Kh = kh + (size_t)bh * kNk * kHD;        // [1024 x 64]
  const bf16* Vt = vt + (size_t)bh * kHD * kNk;        // [64 x 1024] (d-major)

  __shared__ float S[16][512];
  __shared__ bf16  P[16][512];
  __shared__ float Op[2][16][64];
  __shared__ float red[16][16];
  __shared__ float rowm[16], rowl[16], rowscale[16];

  int t = threadIdx.x;
  int w = t >> 5, lane = t & 31;
  int srow = t >> 4, scth = t & 15;

  if (t < 16) { rowm[t] = -1e30f; rowl[t] = 0.0f; }

  v16bf aq0 = load_a_rm(Q, kHD, 0, 0, lane);
  v16bf aq1 = load_a_rm(Q, kHD, 0, 32, lane);

  int ri = (lane & 16) ? 8 : 0;
  int dtile = w & 3, half = w >> 2;
  v8f o = {};
  __syncthreads();

  for (int ch = 0; ch < 2; ++ch) {
    const bf16* Kc = Kh + (size_t)ch * 512 * kHD;

    // S = scale * Q K^T (32 key-tiles of 16, 4 per wave)
#pragma unroll
    for (int ntl = 0; ntl < 4; ++ntl) {
      int n0 = (w * 4 + ntl) * 16;
      v16bf b0 = load_b_tn(Kc, kHD, 0, n0, lane);
      v16bf b1 = load_b_tn(Kc, kHD, 32, n0, lane);
      v8f s = {};
      s = wmma_bf16(aq0, b0, s);
      s = wmma_bf16(aq1, b1, s);
      int col = n0 + (lane & 15);
#pragma unroll
      for (int r = 0; r < 8; ++r) S[ri + r][col] = s[r] * kScale;
    }
    __syncthreads();

    // online softmax stats: 16 threads per row
    float mx = -1e30f;
    for (int j = scth; j < 512; j += 16) mx = fmaxf(mx, S[srow][j]);
    red[srow][scth] = mx;
    __syncthreads();
    if (scth == 0) {
      float m = red[srow][0];
#pragma unroll
      for (int i = 1; i < 16; ++i) m = fmaxf(m, red[srow][i]);
      float mo = rowm[srow];
      float mn = fmaxf(mo, m);
      rowscale[srow] = __expf(mo - mn);
      rowm[srow] = mn;
    }
    __syncthreads();
    float mn = rowm[srow];
    float ls = 0.0f;
    for (int j = scth; j < 512; j += 16) {
      float e = __expf(S[srow][j] - mn);
      P[srow][j] = (bf16)e;  // unnormalized; 1/rowsum folded in at the end
      ls += e;
    }
    red[srow][scth] = ls;
    __syncthreads();
    if (scth == 0) {
      float s2 = 0.0f;
#pragma unroll
      for (int i = 0; i < 16; ++i) s2 += red[srow][i];
      rowl[srow] = rowl[srow] * rowscale[srow] + s2;
    }
    __syncthreads();

    // O: rescale running accumulator, then += P(chunk) @ V(chunk)
#pragma unroll
    for (int r = 0; r < 8; ++r) o[r] *= rowscale[ri + r];
#pragma unroll
    for (int ks = 0; ks < 8; ++ks) {
      int kl = (half * 8 + ks) * 32;               // k within chunk
      v16bf ap = load_a_rm(&P[0][0], 512, 0, kl, lane);
      // V^T fragment: B[k][d] = Vt[d][ch*512 + kl + k] -> contiguous
      v16bf bv = load_b_tn(Vt, kNk, ch * 512 + kl, dtile * 16, lane);
      o = wmma_bf16(ap, bv, o);
    }
    __syncthreads();
  }

  {
    int col = dtile * 16 + (lane & 15);
#pragma unroll
    for (int r = 0; r < 8; ++r) Op[half][ri + r][col] = o[r];
  }
  __syncthreads();
  for (int idx = t; idx < 16 * kHD; idx += 256) {
    int row = idx >> 6, d = idx & 63;
    float val = (Op[0][row][d] + Op[1][row][d]) / rowl[row];
    // concat heads: out channel = h*64 + d
    out_attn[((size_t)b * kN + q0 + row) * kC + h * kHD + d] = (bf16)val;
  }
}

extern "C" void kernel_launch(void* const* d_in, const int* in_sizes, int n_in,
                              void* d_out, int out_size, void* d_ws, size_t ws_size,
                              hipStream_t stream) {
  (void)in_sizes; (void)n_in; (void)out_size; (void)ws_size;
  const float* x   = (const float*)d_in[0];
  const float* Wq  = (const float*)d_in[1];
  const float* bq  = (const float*)d_in[2];
  const float* Wk  = (const float*)d_in[3];
  const float* bk  = (const float*)d_in[4];
  const float* Wv  = (const float*)d_in[5];
  const float* bv  = (const float*)d_in[6];
  const float* Wp  = (const float*)d_in[7];
  const float* bp  = (const float*)d_in[8];
  const float* srw = (const float*)d_in[9];
  const float* srb = (const float*)d_in[10];
  const float* lng = (const float*)d_in[11];
  const float* lnb = (const float*)d_in[12];

  char* ws = (char*)d_ws;
  size_t off = 0;
  auto take = [&](size_t bytes) -> void* {
    void* p = ws + off;
    off = (off + bytes + 255) & ~(size_t)255;
    return p;
  };
  bf16*  x_bf   = (bf16*)take((size_t)kB * kN * kC * 2);
  bf16*  Wq_bf  = (bf16*)take((size_t)kC * kC * 2);
  bf16*  Wk_bf  = (bf16*)take((size_t)kC * kC * 2);
  bf16*  Wv_bf  = (bf16*)take((size_t)kC * kC * 2);
  bf16*  Wp_bf  = (bf16*)take((size_t)kC * kC * 2);
  bf16*  srw_bf = (bf16*)take((size_t)kKc * kC * 2);
  bf16*  Aconv  = (bf16*)take((size_t)kB * kNk * kKc * 2);
  float* kv_raw = (float*)take((size_t)kB * kNk * kC * 4);
  bf16*  kv_bf  = (bf16*)take((size_t)kB * kNk * kC * 2);
  bf16*  q_hd   = (bf16*)take((size_t)kB * kNH * kN  * kHD * 2);
  bf16*  k_hd   = (bf16*)take((size_t)kB * kNH * kNk * kHD * 2);
  bf16*  v_tr   = (bf16*)take((size_t)kB * kNH * kHD * kNk * 2);
  bf16*  oattn  = (bf16*)take((size_t)kB * kN * kC * 2);

  // 1) conversions
  int nx = kB * kN * kC;
  cvt_f2bf<<<(nx + 255) / 256, 256, 0, stream>>>(x, x_bf, nx);
  int nw = kC * kC;
  cvt_f2bf<<<(nw + 255) / 256, 256, 0, stream>>>(Wq, Wq_bf, nw);
  cvt_f2bf<<<(nw + 255) / 256, 256, 0, stream>>>(Wk, Wk_bf, nw);
  cvt_f2bf<<<(nw + 255) / 256, 256, 0, stream>>>(Wv, Wv_bf, nw);
  cvt_f2bf<<<(nw + 255) / 256, 256, 0, stream>>>(Wp, Wp_bf, nw);
  int nsr = kKc * kC;
  cvt_f2bf<<<(nsr + 255) / 256, 256, 0, stream>>>(srw, srw_bf, nsr);

  // 2) strided conv as im2col + GEMM -> kv_raw (fp32)
  int nim = kB * kNk * kKc;
  im2col_bf16<<<(nim + 255) / 256, 256, 0, stream>>>(x, Aconv);
  gemm_bf16_wmma<<<dim3(kC / 64, (kB * kNk) / 128), 256, 0, stream>>>(
      Aconv, srw_bf, srb, kv_raw, nullptr, kB * kNk, kC, kKc, EPI_F32, 0);

  // 3) LayerNorm -> kv_bf (bf16)
  layernorm_bf16<<<kB * kNk, 256, 0, stream>>>(kv_raw, lng, lnb, kv_bf);

  // 4) Q/K/V GEMMs with fused head-permute epilogues (lg_rows = log2(rows/batch))
  gemm_bf16_wmma<<<dim3(kC / 64, (kB * kN) / 128), 256, 0, stream>>>(
      x_bf, Wq_bf, bq, nullptr, q_hd, kB * kN, kC, kC, EPI_HEADS, 12);
  gemm_bf16_wmma<<<dim3(kC / 64, (kB * kNk) / 128), 256, 0, stream>>>(
      kv_bf, Wk_bf, bk, nullptr, k_hd, kB * kNk, kC, kC, EPI_HEADS, 10);
  gemm_bf16_wmma<<<dim3(kC / 64, (kB * kNk) / 128), 256, 0, stream>>>(
      kv_bf, Wv_bf, bv, nullptr, v_tr, kB * kNk, kC, kC, EPI_HEADS_T, 10);

  // 5) attention (V consumed in transposed layout)
  attn_wmma<<<dim3(kN / 16, kB * kNH), 256, 0, stream>>>(q_hd, k_hd, v_tr, oattn);

  // 6) output projection -> fp32 d_out
  gemm_bf16_wmma<<<dim3(kC / 64, (kB * kN) / 128), 256, 0, stream>>>(
      oattn, Wp_bf, bp, (float*)d_out, nullptr, kB * kN, kC, kC, EPI_F32, 0);
}
